// SelfAttentionHead_71193377898910
// MI455X (gfx1250) — matrix operand order repaired and checked by reading, compiled, not run on hardware
//
#include <hip/hip_runtime.h>

typedef __bf16 v16bf __attribute__((ext_vector_type(16)));
typedef __bf16 v8bf  __attribute__((ext_vector_type(8)));
typedef float  v8f   __attribute__((ext_vector_type(8)));
typedef unsigned int u32x4 __attribute__((ext_vector_type(4)));
typedef int i32x4 __attribute__((ext_vector_type(4)));
typedef int i32x8 __attribute__((ext_vector_type(8)));

#define WMMA_BF16(A, B, C) \
  __builtin_amdgcn_wmma_f32_16x16x32_bf16(false, (A), false, (B), (short)0, (C), false, false)

constexpr int Bn  = 4;
constexpr int Tn  = 2048;
constexpr int Cn  = 1024;
constexpr int NH  = 16;
constexpr int HSn = 64;

// TDM builtin exists on both toolchains but with different arity:
//   ROCm 7.2 (clang-22): 5 args;  amdgpu-toolchain (clang-23): 6 args.
#if defined(__HIP_DEVICE_COMPILE__)
#if __has_builtin(__builtin_amdgcn_tensor_load_to_lds)
#define HAVE_TDM 1
#endif
#endif

// Combine two 16-byte chunks into a 16-element bf16 fragment register set.
__device__ __forceinline__ v16bf ld_pair(const __bf16* lo, const __bf16* hi) {
  v8bf a = *(const v8bf*)lo;
  v8bf b = *(const v8bf*)hi;
  return __builtin_shufflevector(a, b, 0, 1, 2, 3, 4, 5, 6, 7, 8, 9, 10, 11, 12, 13, 14, 15);
}

// ---------------------------------------------------------------------------
// f32 -> bf16 conversion, 8 elements / thread
// ---------------------------------------------------------------------------
__global__ __launch_bounds__(256) void cvt_f32_bf16(const float* __restrict__ src,
                                                    __bf16* __restrict__ dst, int n8) {
  int i = blockIdx.x * blockDim.x + threadIdx.x;
  if (i >= n8) return;
  const float4* s = (const float4*)src + 2 * (size_t)i;
  float4 a = s[0];
  float4 c = s[1];
  v8bf o;
  o[0] = (__bf16)a.x; o[1] = (__bf16)a.y; o[2] = (__bf16)a.z; o[3] = (__bf16)a.w;
  o[4] = (__bf16)c.x; o[5] = (__bf16)c.y; o[6] = (__bf16)c.z; o[7] = (__bf16)c.w;
  *((v8bf*)dst + i) = o;
}

// ---------------------------------------------------------------------------
// QKV projection: attn[m][n] = sum_k xb[m][k] * Wb[n][k] + bias[n]
// M=8192, N=3072, K=1024.  Per wave: 32(M) x 64(N) tile (two M sub-tiles
// share each B fragment).  Ping-pong double buffering (two named buffer
// sets, no register rotation copies): each half-iteration loads one set
// while the WMMAs consume the other, so load latency overlaps the matrix
// pipe with only partial s_wait_loadcnt.
// ---------------------------------------------------------------------------
__global__ __launch_bounds__(256) void qkv_proj(const __bf16* __restrict__ xb,
                                                const __bf16* __restrict__ Wb,
                                                const float* __restrict__ bias,
                                                __bf16* __restrict__ Qs,
                                                __bf16* __restrict__ Ks,
                                                __bf16* __restrict__ Vt) {
  const int lane = threadIdx.x & 31;
  const int wave = threadIdx.x >> 5;
  const int lgrp = lane >> 4;   // 0 or 1
  const int ln   = lane & 15;
  const int w    = blockIdx.x * 8 + wave;   // global wave id
  const int mt   = w / 48;                  // M tile (32 rows), 0..255
  const int nb   = w % 48;                  // N block (64 cols), 0..47

  const __bf16* arow0 = xb + (size_t)(mt * 32 + ln) * Cn + lgrp * 8;
  const __bf16* arow1 = arow0 + (size_t)16 * Cn;
  const __bf16* wbase = Wb + (size_t)(nb * 64 + ln) * Cn + lgrp * 8;

  v8f acc[8] = {};

  // Prologue: buffer set A holds k = 0
  v16bf aA0 = ld_pair(arow0, arow0 + 16);
  v16bf aA1 = ld_pair(arow1, arow1 + 16);
  v16bf bA[4];
#pragma unroll
  for (int t = 0; t < 4; ++t) bA[t] = ld_pair(wbase + (size_t)t * 16 * Cn, wbase + (size_t)t * 16 * Cn + 16);

  for (int k0 = 0; k0 < Cn; k0 += 64) {
    const int k1 = k0 + 32;                 // always < Cn
    // load set B (k1) while consuming set A (k0)
    v16bf aB0 = ld_pair(arow0 + k1, arow0 + k1 + 16);
    v16bf aB1 = ld_pair(arow1 + k1, arow1 + k1 + 16);
    v16bf bB[4];
#pragma unroll
    for (int t = 0; t < 4; ++t) bB[t] = ld_pair(wbase + (size_t)t * 16 * Cn + k1, wbase + (size_t)t * 16 * Cn + k1 + 16);
#pragma unroll
    for (int t = 0; t < 4; ++t) acc[t] = WMMA_BF16(aA0, bA[t], acc[t]);
#pragma unroll
    for (int t = 0; t < 4; ++t) acc[4 + t] = WMMA_BF16(aA1, bA[t], acc[4 + t]);

    const int k2 = (k0 + 64) & (Cn - 1);    // wraps on last iter (harmless)
    // refill set A (k2) while consuming set B (k1)
    aA0 = ld_pair(arow0 + k2, arow0 + k2 + 16);
    aA1 = ld_pair(arow1 + k2, arow1 + k2 + 16);
#pragma unroll
    for (int t = 0; t < 4; ++t) bA[t] = ld_pair(wbase + (size_t)t * 16 * Cn + k2, wbase + (size_t)t * 16 * Cn + k2 + 16);
#pragma unroll
    for (int t = 0; t < 4; ++t) acc[t] = WMMA_BF16(aB0, bB[t], acc[t]);
#pragma unroll
    for (int t = 0; t < 4; ++t) acc[4 + t] = WMMA_BF16(aB1, bB[t], acc[4 + t]);
  }

  // Epilogue: bias + scatter with bf16 conversion.
#pragma unroll
  for (int mi = 0; mi < 2; ++mi) {
#pragma unroll
    for (int t = 0; t < 4; ++t) {
      const int n   = nb * 64 + t * 16 + ln;
      const float bv = bias[n];
      const int sel = n >> 10;       // 0=Q, 1=K, 2=V
      const int c   = n & 1023;
      const int h   = c >> 6;
      const int d   = c & 63;
#pragma unroll
      for (int r = 0; r < 8; ++r) {
        const int m    = mt * 32 + mi * 16 + r + 8 * lgrp;   // global row
        const int bidx = m >> 11;
        const int tpos = m & 2047;
        const float v  = acc[mi * 4 + t][r] + bv;
        const size_t bhh = (size_t)bidx * NH + h;
        if (sel == 0) {
          // fold attention scale (1/sqrt(64)) and log2(e) into Q
          Qs[(bhh * Tn + tpos) * HSn + d] = (__bf16)(v * 0.18033688011112042f);
        } else if (sel == 1) {
          Ks[(bhh * Tn + tpos) * HSn + d] = (__bf16)v;
        } else {
          Vt[(bhh * HSn + d) * Tn + tpos] = (__bf16)v;
        }
      }
    }
  }
}

// ---------------------------------------------------------------------------
// Flash attention, causal. Block = 4 waves, each wave owns a 16-row q tile.
// Q tile (contiguous 2KB) staged into LDS by the Tensor Data Mover when
// available; K/V fragments loaded from global (all of K/V fits in L2).
// V fragments are issued together with K fragments so their latency hides
// under the softmax VALU stretch.
// ---------------------------------------------------------------------------
__global__ __launch_bounds__(128) void flash_attn(const __bf16* __restrict__ Qs,
                                                  const __bf16* __restrict__ Ks,
                                                  const __bf16* __restrict__ Vt,
                                                  float* __restrict__ out) {
  __shared__ __align__(16) __bf16 plds[4][16 * 32];
#if defined(HAVE_TDM)
  __shared__ __align__(16) __bf16 qtile[4][16 * HSn];   // 2KB per wave
#endif

  const int lane = threadIdx.x & 31;
  const int wave = threadIdx.x >> 5;
  const int lgrp = lane >> 4;
  const int ln   = lane & 15;
  const int bh   = blockIdx.y;         // b*16 + h
  const int bb   = bh >> 4;
  const int hh   = bh & 15;
  const int q0   = blockIdx.x * 64 + wave * 16;

  const __bf16* Qb = Qs + (size_t)bh * Tn * HSn;
  const __bf16* Kb = Ks + (size_t)bh * Tn * HSn;
  const __bf16* Vb = Vt + (size_t)bh * HSn * Tn;
  __bf16* myP = &plds[wave][0];

  // ---- Resident Q fragments (16 rows x 64 dims -> two 16x32 A fragments) ---
  v16bf qa0, qa1;
#if defined(HAVE_TDM)
  {
    // TDM: DMA the wave's contiguous 16x64 bf16 Q tile into LDS.
    const __bf16* gq = Qb + (size_t)q0 * HSn;
    const unsigned long long ga = (unsigned long long)(uintptr_t)gq;
    const unsigned ga_lo  = __builtin_amdgcn_readfirstlane((unsigned)ga);
    const unsigned ga_hi  = __builtin_amdgcn_readfirstlane((unsigned)(ga >> 32));
    const unsigned ldsoff = __builtin_amdgcn_readfirstlane(
        (unsigned)(uintptr_t)(void*)&qtile[wave][0]);
    // D# group 0: count=1 | lds_addr | global_addr[56:0] | type=2
    u32x4 g0v;
    g0v[0] = 1u;
    g0v[1] = ldsoff;
    g0v[2] = ga_lo;
    g0v[3] = (ga_hi & 0x01FFFFFFu) | (2u << 30);
    // D# group 1: data_size=2B; tensor_dim0=1024, tensor_dim1=1,
    // tile_dim0=1024 (1D copy of 2KB), tensor_dim0_stride=1024.
    i32x8 g1v;
    g1v[0] = 0x00010000;          // data_size=1 (2 bytes), mask=0
    g1v[1] = (int)(1024u << 16);  // tensor_dim0[15:0]
    g1v[2] = 0x00010000;          // tensor_dim0[31:16]=0, tensor_dim1[15:0]=1
    g1v[3] = (int)(1024u << 16);  // tensor_dim1[31:16]=0, tile_dim0=1024
    g1v[4] = 0;                   // tile_dim1=0, tile_dim2=0 (unused)
    g1v[5] = 1024;                // tensor_dim0_stride[31:0]
    g1v[6] = 0;
    g1v[7] = 0;
    i32x4 z4 = {0, 0, 0, 0};
#if (__clang_major__ >= 23)
    i32x8 z8 = {0, 0, 0, 0, 0, 0, 0, 0};
    __builtin_amdgcn_tensor_load_to_lds(g0v, g1v, z4, z4, z8, 0);
#else
    __builtin_amdgcn_tensor_load_to_lds(g0v, g1v, z4, z4, 0);
#endif
    asm volatile("s_wait_tensorcnt 0" ::: "memory");
    const __bf16* qrow = &qtile[wave][0] + (size_t)ln * HSn + lgrp * 8;
    qa0 = ld_pair(qrow, qrow + 16);
    qa1 = ld_pair(qrow + 32, qrow + 48);
  }
#else
  {
    const __bf16* qrow = Qb + (size_t)(q0 + ln) * HSn + lgrp * 8;
    qa0 = ld_pair(qrow, qrow + 16);
    qa1 = ld_pair(qrow + 32, qrow + 48);
  }
#endif

  v8f o0 = {}, o1 = {}, o2 = {}, o3 = {};
  float mrow[8], lrow[8];
#pragma unroll
  for (int r = 0; r < 8; ++r) { mrow[r] = -__builtin_inff(); lrow[r] = 0.0f; }

  for (int kb = 0; kb < q0 + 16; kb += 32) {
    if (kb + 32 < q0 + 16) {
      __builtin_prefetch(Kb + (size_t)(kb + 32 + ln) * HSn, 0, 3);
      __builtin_prefetch(Vb + (size_t)ln * Tn + kb + 32, 0, 3);
    }

    // Issue ALL global fragment loads for this key block up front:
    // K fragments feed the S WMMAs now; V fragments are consumed only after
    // softmax, so their latency hides under ~100 VALU ops.
    const __bf16* k0p = Kb + (size_t)(kb + ln) * HSn + lgrp * 8;
    const __bf16* k1p = Kb + (size_t)(kb + 16 + ln) * HSn + lgrp * 8;
    const __bf16* v0p = Vb + (size_t)ln * Tn + kb + lgrp * 8;
    v16bf kf0a = ld_pair(k0p, k0p + 16);
    v16bf kf0b = ld_pair(k0p + 32, k0p + 48);
    v16bf kf1a = ld_pair(k1p, k1p + 16);
    v16bf kf1b = ld_pair(k1p + 32, k1p + 48);
    v16bf vf0  = ld_pair(v0p + (size_t)0 * 16 * Tn, v0p + (size_t)0 * 16 * Tn + 16);
    v16bf vf1  = ld_pair(v0p + (size_t)1 * 16 * Tn, v0p + (size_t)1 * 16 * Tn + 16);
    v16bf vf2  = ld_pair(v0p + (size_t)2 * 16 * Tn, v0p + (size_t)2 * 16 * Tn + 16);
    v16bf vf3  = ld_pair(v0p + (size_t)3 * 16 * Tn, v0p + (size_t)3 * 16 * Tn + 16);

    // S = Q * K^T  (two 16x16 tiles over this 32-key block)
    v8f s0 = {}, s1 = {};
    s0 = WMMA_BF16(qa0, kf0a, s0);
    s0 = WMMA_BF16(qa1, kf0b, s0);
    s1 = WMMA_BF16(qa0, kf1a, s1);
    s1 = WMMA_BF16(qa1, kf1b, s1);

    const bool full = (kb + 31) <= q0;   // whole block below the diagonal
    const int c0 = kb + ln;
    const int c1 = c0 + 16;

#pragma unroll
    for (int r = 0; r < 8; ++r) {
      const int row = q0 + r + 8 * lgrp;
      float v0 = s0[r];
      float v1 = s1[r];
      if (!full) {
        if (c0 > row) v0 = -__builtin_inff();
        if (c1 > row) v1 = -__builtin_inff();
      }
      // row max across the 16 lanes holding this row
      float mx = fmaxf(v0, v1);
      mx = fmaxf(mx, __shfl_xor(mx, 8, 16));
      mx = fmaxf(mx, __shfl_xor(mx, 4, 16));
      mx = fmaxf(mx, __shfl_xor(mx, 2, 16));
      mx = fmaxf(mx, __shfl_xor(mx, 1, 16));
      const float mn   = fmaxf(mrow[r], mx);
      const float corr = exp2f(mrow[r] - mn);
      const float p0   = exp2f(v0 - mn);
      const float p1   = exp2f(v1 - mn);
      float rs = p0 + p1;
      rs += __shfl_xor(rs, 8, 16);
      rs += __shfl_xor(rs, 4, 16);
      rs += __shfl_xor(rs, 2, 16);
      rs += __shfl_xor(rs, 1, 16);
      lrow[r] = lrow[r] * corr + rs;
      mrow[r] = mn;
      o0[r] *= corr; o1[r] *= corr; o2[r] *= corr; o3[r] *= corr;
      // stage P tile (C layout -> row-major 16x32 bf16 in LDS)
      const int lr = r + 8 * lgrp;
      myP[lr * 32 + ln]      = (__bf16)p0;
      myP[lr * 32 + 16 + ln] = (__bf16)p1;
    }

    // DS ops from one wave are processed in order; fence the compiler too.
    asm volatile("s_wait_dscnt 0" ::: "memory");

    // Reload P as a 16x32 A fragment.
    v16bf pa;
    {
      const __bf16* prow = myP + ln * 32 + lgrp * 8;
      pa = ld_pair(prow, prow + 16);
    }

    // O += P * V (fragments already resident in registers)
    o0 = WMMA_BF16(pa, vf0, o0);
    o1 = WMMA_BF16(pa, vf1, o1);
    o2 = WMMA_BF16(pa, vf2, o2);
    o3 = WMMA_BF16(pa, vf3, o3);
  }

  // Normalize and store f32 output [B, T, C] with C = h*64 + d.
  float* obase = out + (size_t)bb * Tn * Cn + hh * HSn;
#pragma unroll
  for (int r = 0; r < 8; ++r) {
    const int row   = q0 + r + 8 * lgrp;
    const float inv = 1.0f / lrow[r];
    float* orow = obase + (size_t)row * Cn;
    orow[ln]      = o0[r] * inv;
    orow[16 + ln] = o1[r] * inv;
    orow[32 + ln] = o2[r] * inv;
    orow[48 + ln] = o3[r] * inv;
  }
}

// ---------------------------------------------------------------------------
// Host launcher
// ---------------------------------------------------------------------------
extern "C" void kernel_launch(void* const* d_in, const int* in_sizes, int n_in,
                              void* d_out, int out_size, void* d_ws, size_t ws_size,
                              hipStream_t stream) {
  const float* x    = (const float*)d_in[0];   // [B, T, C]
  const float* W    = (const float*)d_in[1];   // [3C, C]
  const float* bias = (const float*)d_in[2];   // [3C]
  float* out = (float*)d_out;                  // [B, T, C]

  char* ws = (char*)d_ws;
  // Workspace layout (all 16B aligned):
  //   xb : B*T*C bf16        = 16 MiB  @ 0
  //   Wb : 3C*C bf16         =  6 MiB  @ 33554432
  //   Qs : B*nh*T*hs bf16    = 16 MiB  @ 39845888
  //   Ks : B*nh*T*hs bf16    = 16 MiB  @ 56623104
  //   Vt : B*nh*hs*T bf16    = 16 MiB  @ 73400320   (total ~86 MiB)
  __bf16* xb = (__bf16*)(ws);
  __bf16* Wb = (__bf16*)(ws + (size_t)33554432);
  __bf16* Qs = (__bf16*)(ws + (size_t)39845888);
  __bf16* Ks = (__bf16*)(ws + (size_t)56623104);
  __bf16* Vt = (__bf16*)(ws + (size_t)73400320);

  cvt_f32_bf16<<<4096, 256, 0, stream>>>(x, xb, (Bn * Tn * Cn) / 8);
  cvt_f32_bf16<<<1536, 256, 0, stream>>>(W, Wb, (3 * Cn * Cn) / 8);

  // 256 M-tiles (32 rows) x 48 N-blocks = 12288 waves, 8 waves / block
  qkv_proj<<<1536, 256, 0, stream>>>(xb, Wb, bias, Qs, Ks, Vt);

  // grid: 32 q-blocks (64 rows each) x 64 (b,h) pairs, 4 waves / block
  flash_attn<<<dim3(32, 64), 128, 0, stream>>>(Qs, Ks, Vt, out);
}
